// Crf_82695300317883
// MI455X (gfx1250) — matrix operand order repaired and checked by reading, not compiled, not run on hardware
//
#include <hip/hip_runtime.h>

typedef __attribute__((ext_vector_type(16))) _Float16 v16h;
typedef __attribute__((ext_vector_type(8)))  _Float16 v8h;
typedef __attribute__((ext_vector_type(8)))  float    v8f;

// Problem sizes (from the reference)
#define S_LEN 512
#define B_SZ  128
#define T_SZ  256
#define A_STRIDE 264   // halves; 264*2B=528B row stride -> rows hit distinct LDS banks

// ---------------------------------------------------------------------------
// Kernel 1: per-column max of transitions and PT[n][k] = exp(trans[k][n]-Mt[n])
// stored transposed in f16 so WMMA B-fragments are contiguous-in-K.
// ---------------------------------------------------------------------------
__global__ __launch_bounds__(256) void crf_prep(const float* __restrict__ trans,
                                                _Float16* __restrict__ PT,
                                                float* __restrict__ Mt) {
    __shared__ float red[256];
    const int n = blockIdx.x;      // column of transitions
    const int i = threadIdx.x;     // row of transitions
    const float v = trans[(size_t)i * T_SZ + n];
    red[i] = v;
    __syncthreads();
    for (int off = 128; off > 0; off >>= 1) {
        if (i < off) red[i] = fmaxf(red[i], red[i + off]);
        __syncthreads();
    }
    const float mt = red[0];
    if (i == 0) Mt[n] = mt;
    PT[(size_t)n * T_SZ + i] = (_Float16)__expf(v - mt);
}

// ---------------------------------------------------------------------------
// Kernel 2: sequential CRF forward scan. One block = 16 batch rows, 8 waves.
// Per step: row-max -> exp panel (f16, LDS) -> 16x256 = (16x256)x(256x256)
// via v_wmma_f32_16x16x32_f16 -> fused feat + ms + Mt + log(C) writeback.
// ---------------------------------------------------------------------------
__global__ __launch_bounds__(256) void crf_scan(const float* __restrict__ feats,
                                                const int*   __restrict__ mask,
                                                const float* __restrict__ startT,
                                                const _Float16* __restrict__ PT,
                                                const float* __restrict__ Mt,
                                                const float* __restrict__ endT,
                                                float* __restrict__ fs) {
    __shared__ float    scoreLds[16 * T_SZ];        // 16 KB  (f32 state)
    __shared__ _Float16 aLds[16 * A_STRIDE];        // 8.25 KB (exp panel, padded)
    __shared__ float    ms[16];
    __shared__ float    red[16][17];
    __shared__ int      mrow[16];
    __shared__ float    mtLds[T_SZ];

    const int tid   = threadIdx.x;
    const int bBase = blockIdx.x * 16;              // batch chunk
    const int lane  = tid & 31;
    const int wave  = tid >> 5;                     // 0..7
    const int laneM = lane & 15;
    const int hi    = lane >> 4;                    // 0/1: K-half selector
    const int row   = tid >> 4;                     // 0..15
    const int sub   = tid & 15;                     // 0..15

    mtLds[tid] = Mt[tid];                           // T_SZ == blockDim.x

    // score0 = start_transitions + feats[0]
    {
        const float* f0 = feats + (size_t)(bBase + row) * T_SZ;
        const int c0 = sub * 16;
        #pragma unroll 4
        for (int q = 0; q < 16; ++q)
            scoreLds[row * T_SZ + c0 + q] = startT[c0 + q] + f0[c0 + q];
    }
    __syncthreads();

    for (int t = 1; t < S_LEN; ++t) {
        // ---- stage A: per-row max over T (16 threads/row), load step mask ----
        float pm = -3.4e38f;
        #pragma unroll 4
        for (int q = 0; q < 16; ++q)
            pm = fmaxf(pm, scoreLds[row * T_SZ + sub * 16 + q]);
        red[row][sub] = pm;
        if (tid < 16) mrow[tid] = mask[t * B_SZ + bBase + tid];
        __syncthreads();
        if (sub == 0) {
            float mm = red[row][0];
            #pragma unroll
            for (int q = 1; q < 16; ++q) mm = fmaxf(mm, red[row][q]);
            ms[row] = mm;
        }
        __syncthreads();

        // ---- stage B: A = exp(score - ms) into padded f16 LDS panel ----
        {
            const float mr = ms[row];
            #pragma unroll 4
            for (int q = 0; q < 16; ++q)
                aLds[row * A_STRIDE + sub * 16 + q] =
                    (_Float16)__expf(scoreLds[row * T_SZ + sub * 16 + q] - mr);
        }
        __syncthreads();

        // ---- stage C: WMMA GEMM + fused log writeback ----
        #pragma unroll
        for (int tile = 0; tile < 2; ++tile) {
            const int nBase = (wave * 2 + tile) * 16;
            const int j = nBase + laneM;                  // output column
            const _Float16* aRow = aLds + laneM * A_STRIDE;
            const _Float16* bRow = PT + (size_t)j * T_SZ; // PT[n][k], contiguous K
            v8f c = {};
            #pragma unroll
            for (int kb = 0; kb < T_SZ; kb += 32) {
                // A 16x32 f16 frag: lane holds row laneM, K 0..15 / 16..31 by hi
                v8h a0 = *(const v8h*)(aRow + kb + hi * 16);
                v8h a1 = *(const v8h*)(aRow + kb + hi * 16 + 8);
                // B 32x16 f16 frag: lane holds column j, K 0..15 / 16..31 by hi
                v8h b0 = *(const v8h*)(bRow + kb + hi * 16);
                v8h b1 = *(const v8h*)(bRow + kb + hi * 16 + 8);
                v16h a = __builtin_shufflevector(a0, a1, 0,1,2,3,4,5,6,7,8,9,10,11,12,13,14,15);
                v16h b = __builtin_shufflevector(b0, b1, 0,1,2,3,4,5,6,7,8,9,10,11,12,13,14,15);
                c = __builtin_amdgcn_wmma_f32_16x16x32_f16(
                        false, a, false, b, (short)0, c, false, false);
            }
            // C/D layout: VGPR r -> (M = r + 8*hi, N = laneM)
            const float mtj = mtLds[j];
            const float* fcol = feats + ((size_t)t * B_SZ + bBase) * T_SZ + j;
            #pragma unroll
            for (int r = 0; r < 8; ++r) {
                const int mr = r + hi * 8;
                const float nv = fcol[mr * T_SZ] + ms[mr] + mtj
                               + __logf(fmaxf(c[r], 1e-30f));
                if (mrow[mr]) scoreLds[mr * T_SZ + j] = nv;   // mask: keep old if 0
            }
        }
        if (t + 1 < S_LEN) {   // warm next step's emissions into cache
            __builtin_prefetch(feats + ((size_t)(t + 1) * B_SZ + bBase + row) * T_SZ + sub * 16, 0, 1);
        }
        __syncthreads();
    }

    // ---- tail: forward_score[b] = logsumexp_j(score[b][j] + end[j]) ----
    float pm = -3.4e38f;
    #pragma unroll 4
    for (int q = 0; q < 16; ++q)
        pm = fmaxf(pm, scoreLds[row * T_SZ + sub * 16 + q] + endT[sub * 16 + q]);
    red[row][sub] = pm;
    __syncthreads();
    if (sub == 0) {
        float mm = red[row][0];
        #pragma unroll
        for (int q = 1; q < 16; ++q) mm = fmaxf(mm, red[row][q]);
        ms[row] = mm;
    }
    __syncthreads();
    float s = 0.f;
    #pragma unroll 4
    for (int q = 0; q < 16; ++q)
        s += __expf(scoreLds[row * T_SZ + sub * 16 + q] + endT[sub * 16 + q] - ms[row]);
    red[row][sub] = s;
    __syncthreads();
    if (sub == 0) {
        float ss = 0.f;
        #pragma unroll
        for (int q = 0; q < 16; ++q) ss += red[row][q];
        fs[bBase + row] = ms[row] + __logf(ss);
    }
}

// ---------------------------------------------------------------------------
// Kernel 3: gold path score (gathers) and final output = forward - gold.
// ---------------------------------------------------------------------------
__global__ __launch_bounds__(128) void crf_gold(const float* __restrict__ feats,
                                                const int*   __restrict__ tags,
                                                const int*   __restrict__ mask,
                                                const float* __restrict__ startT,
                                                const float* __restrict__ endT,
                                                const float* __restrict__ trans,
                                                const float* __restrict__ fs,
                                                float* __restrict__ out) {
    const int b = threadIdx.x;            // B_SZ == 128
    int prev = tags[b];
    float gold = startT[prev] + feats[(size_t)b * T_SZ + prev];
    int cnt = (mask[b] != 0) ? 1 : 0;
    for (int t = 1; t < S_LEN; ++t) {
        const int   tg = tags[t * B_SZ + b];
        const int   mv = mask[t * B_SZ + b];
        if (mv) {
            gold += feats[((size_t)t * B_SZ + b) * T_SZ + tg]
                  + trans[prev * T_SZ + tg];
            ++cnt;
        }
        prev = tg;
    }
    const int lastTag = tags[(cnt - 1) * B_SZ + b];
    gold += endT[lastTag];
    out[b] = fs[b] - gold;
}

// ---------------------------------------------------------------------------
extern "C" void kernel_launch(void* const* d_in, const int* in_sizes, int n_in,
                              void* d_out, int out_size, void* d_ws, size_t ws_size,
                              hipStream_t stream) {
    const float* feats  = (const float*)d_in[0];
    const int*   tags   = (const int*)d_in[1];
    const int*   mask   = (const int*)d_in[2];
    const float* startT = (const float*)d_in[3];
    const float* endT   = (const float*)d_in[4];
    const float* trans  = (const float*)d_in[5];
    float* out = (float*)d_out;

    char* ws = (char*)d_ws;
    _Float16* PT = (_Float16*)ws;                              // 256*256*2 = 128 KB
    float*    Mt = (float*)(ws + (size_t)T_SZ * T_SZ * 2);     // 1 KB
    float*    fs = (float*)(ws + (size_t)T_SZ * T_SZ * 2 + T_SZ * 4); // 512 B

    crf_prep<<<T_SZ, 256, 0, stream>>>(trans, PT, Mt);
    crf_scan<<<B_SZ / 16, 256, 0, stream>>>(feats, mask, startT, PT, Mt, endT, fs);
    crf_gold<<<1, B_SZ, 0, stream>>>(feats, tags, mask, startT, endT, trans, fs, out);
}
